// MHA_80092550136047
// MI455X (gfx1250) — matrix operand order repaired and checked by reading, not compiled, not run on hardware
//
#include <hip/hip_runtime.h>
#include <hip/hip_bf16.h>

// ---------- types ----------
typedef __attribute__((ext_vector_type(16))) _Float16 v16h;
typedef __attribute__((ext_vector_type(8)))  _Float16 h8;
typedef __attribute__((ext_vector_type(4)))  _Float16 h4;
typedef __attribute__((ext_vector_type(8)))  float    v8f;
typedef __attribute__((ext_vector_type(4)))  unsigned int v4u;
typedef __attribute__((ext_vector_type(8)))  int      v8i;
typedef __attribute__((ext_vector_type(4)))  int      v4i;

union FragAB { v16h v; h8 h[2]; };

#define D_DIM 1024
#define S_DIM 1024
#define H_NUM 16
#define HD    64

// ---------- CDNA5 async global->LDS copy (ASYNCcnt-tracked) ----------
__device__ __forceinline__ void async_copy_b128(unsigned lds_off,
                                                unsigned long long gaddr) {
  asm volatile("global_load_async_to_lds_b128 %0, %1, off"
               :
               : "v"(lds_off), "v"(gaddr)
               : "memory");
}
__device__ __forceinline__ void wait_async() {
  asm volatile("s_wait_asynccnt 0x0" ::: "memory");
}

// ---------- CDNA5 TDM: 1-D contiguous tensor tile -> LDS ----------
// nelems f16 elements (contiguous), nelems < 65536 so it fits tile_dim0.
__device__ __forceinline__ void tdm_load_1d(unsigned lds_off, const void* g,
                                            unsigned nelems) {
  unsigned long long ga = (unsigned long long)g;
  v4u g0;
  g0[0] = 1u;                                   // count=1, user mode, no gather
  g0[1] = lds_off;                              // lds_addr
  g0[2] = (unsigned)(ga & 0xffffffffu);         // global_addr[31:0]
  g0[3] = (unsigned)((ga >> 32) & 0x01ffffffu)  // global_addr[56:32]
          | 0x80000000u;                        // type=2 ("image")
  v8i g1;
  g1[0] = (int)(1u << 16);                      // wg_mask=0, data_size=1 (2B)
  g1[1] = (int)((nelems & 0xffffu) << 16);      // tensor_dim0[15:0]
  g1[2] = (int)((nelems >> 16) | (1u << 16));   // tensor_dim0[31:16], tensor_dim1=1
  g1[3] = (int)(nelems << 16);                  // tensor_dim1 hi=0, tile_dim0=nelems
  g1[4] = 1;                                    // tile_dim1=1, tile_dim2=0
  g1[5] = (int)nelems;                          // tensor_dim0_stride[31:0]
  g1[6] = 0;                                    // stride0 hi, stride1 lo
  g1[7] = 0;                                    // stride1 hi
  v4i gz4 = {0, 0, 0, 0};
  v8i gz8 = {0, 0, 0, 0, 0, 0, 0, 0};
  __builtin_amdgcn_tensor_load_to_lds(g0, g1, gz4, gz4, gz8, 0);
}

// ---------- f32 -> f16 conversion ----------
__global__ __launch_bounds__(256) void f32_to_f16_kernel(
    const float* __restrict__ src, _Float16* __restrict__ dst, int n) {
  int i = (blockIdx.x * 256 + threadIdx.x) * 4;
  if (i + 3 < n) {
    float4 v = *(const float4*)(src + i);
    h4 o;
    o[0] = (_Float16)v.x; o[1] = (_Float16)v.y;
    o[2] = (_Float16)v.z; o[3] = (_Float16)v.w;
    *(h4*)(dst + i) = o;
  }
}

// ---------- generic projection GEMM: C = A[4096x1024] @ W[1024x1024] + bias ----------
__global__ __launch_bounds__(256) void gemm_proj_kernel(
    const _Float16* __restrict__ A, const _Float16* __restrict__ W,
    const float* __restrict__ bias, float* __restrict__ outF,
    _Float16* __restrict__ outQKV, float scale) {
  __shared__ __align__(16) _Float16 As[64][32];
  __shared__ __align__(16) _Float16 BsT[128][32];
  const int t = threadIdx.x;
  const int lane = t & 31, w = t >> 5;
  const int wm = w >> 2, wn = w & 3;           // 2 x 4 wave grid
  const int half = lane >> 4, l16 = lane & 15;
  const int m0 = blockIdx.y * 64;
  const int n0 = blockIdx.x * 128;

  v8f acc[2][2] = {};

  for (int k0 = 0; k0 < D_DIM; k0 += 32) {
    __syncthreads();
    {  // A tile 64x32: async global->LDS, one b128 per lane
      const int row = t >> 2, grp = t & 3;
      async_copy_b128(
          (unsigned)(size_t)(&As[row][grp * 8]),
          (unsigned long long)(A + (size_t)(m0 + row) * D_DIM + k0 + grp * 8));
    }
#pragma unroll
    for (int i = 0; i < 2; ++i) {  // B tile 32x128, transposed into LDS
      const int lid = t + 256 * i;
      const int kr = lid >> 4, ng = lid & 15;
      h8 vv = *(const h8*)(W + (size_t)(k0 + kr) * D_DIM + n0 + ng * 8);
      if (k0 + 32 < D_DIM)  // prefetch next B tile row -> global_prefetch_b8
        __builtin_prefetch(W + (size_t)(k0 + 32 + kr) * D_DIM + n0 + ng * 8, 0, 3);
#pragma unroll
      for (int j = 0; j < 8; ++j) BsT[ng * 8 + j][kr] = vv[j];
    }
    wait_async();
    __syncthreads();

    FragAB fa[2], fb[2];
#pragma unroll
    for (int mi = 0; mi < 2; ++mi) {
      const int r = wm * 32 + mi * 16 + l16;
      fa[mi].h[0] = *(const h8*)(&As[r][half * 8]);
      fa[mi].h[1] = *(const h8*)(&As[r][16 + half * 8]);
    }
#pragma unroll
    for (int ni = 0; ni < 2; ++ni) {
      const int c = wn * 32 + ni * 16 + l16;
      fb[ni].h[0] = *(const h8*)(&BsT[c][half * 8]);
      fb[ni].h[1] = *(const h8*)(&BsT[c][16 + half * 8]);
    }
#pragma unroll
    for (int mi = 0; mi < 2; ++mi)
#pragma unroll
      for (int ni = 0; ni < 2; ++ni)
        acc[mi][ni] = __builtin_amdgcn_wmma_f32_16x16x32_f16(
            false, fa[mi].v, false, fb[ni].v, (short)0, acc[mi][ni],
            false, false);
  }

#pragma unroll
  for (int mi = 0; mi < 2; ++mi) {
#pragma unroll
    for (int ni = 0; ni < 2; ++ni) {
      const int col = n0 + wn * 32 + ni * 16 + l16;
      const float bv = bias[col];
#pragma unroll
      for (int v = 0; v < 8; ++v) {
        const int row = m0 + wm * 32 + mi * 16 + half * 8 + v;
        const float val = acc[mi][ni][v] + bv;
        if (outQKV) {
          const int b = row >> 10, s = row & 1023;
          const int h = col >> 6, d = col & 63;
          outQKV[(((size_t)(b * H_NUM + h) * S_DIM) + s) * HD + d] =
              (_Float16)(val * scale);
        } else {
          outF[(size_t)row * D_DIM + col] = val;
        }
      }
    }
  }
}

// ---------- scores = Q @ K^T  (Q pre-scaled by 1/8), fp32 into probs ----------
__global__ __launch_bounds__(256) void attn_scores_kernel(
    const _Float16* __restrict__ Q, const _Float16* __restrict__ Km,
    float* __restrict__ probs) {
  __shared__ __align__(16) _Float16 Qs[64][64];
  __shared__ __align__(16) _Float16 Ks[128][64];
  const int t = threadIdx.x, lane = t & 31, w = t >> 5;
  const int wm = w >> 2, wn = w & 3;
  const int half = lane >> 4, l16 = lane & 15;
  const int bh = blockIdx.z;
  const int m0 = blockIdx.y * 64;
  const int n0 = blockIdx.x * 128;
  const _Float16* Qb = Q + (size_t)bh * S_DIM * HD;
  const _Float16* Kb = Km + (size_t)bh * S_DIM * HD;

  // TDM: stage both tiles (fully contiguous: row stride == tile width).
  if (w == 0) {
    tdm_load_1d((unsigned)(size_t)(&Qs[0][0]), Qb + (size_t)m0 * HD, 64 * HD);
    tdm_load_1d((unsigned)(size_t)(&Ks[0][0]), Kb + (size_t)n0 * HD, 128 * HD);
  }
  __builtin_amdgcn_s_wait_tensorcnt(0);
  __syncthreads();

  v8f acc[2][2] = {};
#pragma unroll
  for (int kk = 0; kk < HD; kk += 32) {
    FragAB fa[2], fb[2];
#pragma unroll
    for (int mi = 0; mi < 2; ++mi) {
      const int r = wm * 32 + mi * 16 + l16;
      fa[mi].h[0] = *(const h8*)(&Qs[r][kk + half * 8]);
      fa[mi].h[1] = *(const h8*)(&Qs[r][kk + 16 + half * 8]);
    }
#pragma unroll
    for (int ni = 0; ni < 2; ++ni) {
      const int c = wn * 32 + ni * 16 + l16;
      fb[ni].h[0] = *(const h8*)(&Ks[c][kk + half * 8]);
      fb[ni].h[1] = *(const h8*)(&Ks[c][kk + 16 + half * 8]);
    }
#pragma unroll
    for (int mi = 0; mi < 2; ++mi)
#pragma unroll
      for (int ni = 0; ni < 2; ++ni)
        acc[mi][ni] = __builtin_amdgcn_wmma_f32_16x16x32_f16(
            false, fa[mi].v, false, fb[ni].v, (short)0, acc[mi][ni],
            false, false);
  }

  float* P = probs + (size_t)bh * S_DIM * S_DIM;
#pragma unroll
  for (int mi = 0; mi < 2; ++mi)
#pragma unroll
    for (int ni = 0; ni < 2; ++ni) {
      const int col = n0 + wn * 32 + ni * 16 + l16;
#pragma unroll
      for (int v = 0; v < 8; ++v) {
        const int row = m0 + wm * 32 + mi * 16 + half * 8 + v;
        P[(size_t)row * S_DIM + col] = acc[mi][ni][v];
      }
    }
}

// ---------- row softmax, in place over probs ----------
__global__ __launch_bounds__(256) void softmax_rows_kernel(float* __restrict__ probs) {
  float* p = probs + (size_t)blockIdx.x * S_DIM;
  const int t = threadIdx.x, lane = t & 31, w = t >> 5;
  __shared__ float red[8];
  float4 v = ((const float4*)p)[t];
  float m = fmaxf(fmaxf(v.x, v.y), fmaxf(v.z, v.w));
#pragma unroll
  for (int off = 16; off; off >>= 1) m = fmaxf(m, __shfl_xor(m, off, 32));
  if (lane == 0) red[w] = m;
  __syncthreads();
#pragma unroll
  for (int i = 0; i < 8; ++i) m = fmaxf(m, red[i]);
  v.x = __expf(v.x - m); v.y = __expf(v.y - m);
  v.z = __expf(v.z - m); v.w = __expf(v.w - m);
  float s = v.x + v.y + v.z + v.w;
#pragma unroll
  for (int off = 16; off; off >>= 1) s += __shfl_xor(s, off, 32);
  __syncthreads();
  if (lane == 0) red[w] = s;
  __syncthreads();
  s = 0.f;
#pragma unroll
  for (int i = 0; i < 8; ++i) s += red[i];
  const float inv = 1.0f / s;
  v.x *= inv; v.y *= inv; v.z *= inv; v.w *= inv;
  ((float4*)p)[t] = v;
}

// ---------- ctx = probs @ V, f16 out in concat-head [B,S,1024] layout ----------
__global__ __launch_bounds__(256) void attn_ctx_kernel(
    const float* __restrict__ probs, const _Float16* __restrict__ V,
    _Float16* __restrict__ ctx) {
  __shared__ __align__(16) _Float16 Ps[64][32];
  __shared__ __align__(16) _Float16 VsT[64][32];
  const int t = threadIdx.x, lane = t & 31, w = t >> 5;
  const int wm = w >> 1, wn = w & 1;  // 4 x 2 wave grid; wave tile 16x32
  const int half = lane >> 4, l16 = lane & 15;
  const int bh = blockIdx.z;
  const int b = bh >> 4, h = bh & 15;
  const int m0 = blockIdx.y * 64;
  const float* Pb = probs + (size_t)bh * S_DIM * S_DIM;
  const _Float16* Vb = V + (size_t)bh * S_DIM * HD;

  v8f acc[2] = {};
  for (int k0 = 0; k0 < S_DIM; k0 += 32) {
    __syncthreads();
    {  // probs tile 64x32, fp32 -> f16
      const int row = t >> 2, grp = t & 3;
      const float* src = Pb + (size_t)(m0 + row) * S_DIM + k0 + grp * 8;
      h8 d;
#pragma unroll
      for (int j = 0; j < 8; ++j) d[j] = (_Float16)src[j];
      *(h8*)(&Ps[row][grp * 8]) = d;
    }
    {  // V tile 32x64, transposed into LDS
      const int kr = t >> 3, ng = t & 7;
      h8 vv = *(const h8*)(Vb + (size_t)(k0 + kr) * HD + ng * 8);
#pragma unroll
      for (int j = 0; j < 8; ++j) VsT[ng * 8 + j][kr] = vv[j];
    }
    __syncthreads();

    FragAB fa, fb[2];
    const int r = wm * 16 + l16;
    fa.h[0] = *(const h8*)(&Ps[r][half * 8]);
    fa.h[1] = *(const h8*)(&Ps[r][16 + half * 8]);
#pragma unroll
    for (int ni = 0; ni < 2; ++ni) {
      const int c = wn * 32 + ni * 16 + l16;
      fb[ni].h[0] = *(const h8*)(&VsT[c][half * 8]);
      fb[ni].h[1] = *(const h8*)(&VsT[c][16 + half * 8]);
    }
#pragma unroll
    for (int ni = 0; ni < 2; ++ni)
      acc[ni] = __builtin_amdgcn_wmma_f32_16x16x32_f16(
          false, fa.v, false, fb[ni].v, (short)0, acc[ni], false, false);
  }

#pragma unroll
  for (int ni = 0; ni < 2; ++ni) {
    const int col = wn * 32 + ni * 16 + l16;  // within-head column
#pragma unroll
    for (int v = 0; v < 8; ++v) {
      const int s = m0 + wm * 16 + half * 8 + v;
      ctx[((size_t)b * S_DIM + s) * D_DIM + h * HD + col] = (_Float16)acc[ni][v];
    }
  }
}

// ---------- host launcher ----------
extern "C" void kernel_launch(void* const* d_in, const int* in_sizes, int n_in,
                              void* d_out, int out_size, void* d_ws, size_t ws_size,
                              hipStream_t stream) {
  (void)in_sizes; (void)n_in; (void)out_size; (void)ws_size;
  const float* x  = (const float*)d_in[0];
  const float* Wq = (const float*)d_in[1];
  const float* bq = (const float*)d_in[2];
  const float* Wk = (const float*)d_in[3];
  const float* bk = (const float*)d_in[4];
  const float* Wv = (const float*)d_in[5];
  const float* bv = (const float*)d_in[6];
  const float* Wo = (const float*)d_in[7];
  const float* bo = (const float*)d_in[8];

  float* out   = (float*)d_out;
  float* probs = out + (size_t)4 * 1024 * 1024;  // out then probs, flat

  const size_t NX = (size_t)4096 * 1024;  // 4M activations
  const size_t NW = (size_t)1024 * 1024;  // 1M weights
  _Float16* Xb  = (_Float16*)d_ws;
  _Float16* Wqb = Xb + NX;
  _Float16* Wkb = Wqb + NW;
  _Float16* Wvb = Wkb + NW;
  _Float16* Wob = Wvb + NW;
  _Float16* Qb  = Wob + NW;   // [B,H,S,64]
  _Float16* Kb  = Qb + NX;
  _Float16* Vb  = Kb + NX;
  _Float16* Cb  = Vb + NX;    // ctx, [B,S,1024]

  f32_to_f16_kernel<<<4096, 256, 0, stream>>>(x,  Xb,  (int)NX);
  f32_to_f16_kernel<<<1024, 256, 0, stream>>>(Wq, Wqb, (int)NW);
  f32_to_f16_kernel<<<1024, 256, 0, stream>>>(Wk, Wkb, (int)NW);
  f32_to_f16_kernel<<<1024, 256, 0, stream>>>(Wv, Wvb, (int)NW);
  f32_to_f16_kernel<<<1024, 256, 0, stream>>>(Wo, Wob, (int)NW);

  dim3 gp(8, 64, 1);  // N/128 x M/64
  gemm_proj_kernel<<<gp, 256, 0, stream>>>(Xb, Wqb, bq, nullptr, Qb, 0.125f);
  gemm_proj_kernel<<<gp, 256, 0, stream>>>(Xb, Wkb, bk, nullptr, Kb, 1.0f);
  gemm_proj_kernel<<<gp, 256, 0, stream>>>(Xb, Wvb, bv, nullptr, Vb, 1.0f);

  attn_scores_kernel<<<dim3(8, 16, 64), 256, 0, stream>>>(Qb, Kb, probs);
  softmax_rows_kernel<<<65536, 256, 0, stream>>>(probs);
  attn_ctx_kernel<<<dim3(1, 16, 64), 256, 0, stream>>>(probs, Vb, Cb);

  gemm_proj_kernel<<<gp, 256, 0, stream>>>(Cb, Wob, bo, out, nullptr, 1.0f);
}
